// NNUE_83640193122666
// MI455X (gfx1250) — matrix operand order repaired and checked by reading, not compiled, not run on hardware
//
#include <hip/hip_runtime.h>

#define INPUT_SIZE 768
#define ACC_SIZE   1024
#define COMBINED   (2 * ACC_SIZE)
#define KT         32                 // K per WMMA step
#define KT2        (KT / 2)           // u32 (bf16-pair) columns
#define BROWS      128                // batch rows per block
#define BCOLS      256                // accumulator cols per block
#define NKT        (INPUT_SIZE / KT)  // 24

typedef __attribute__((ext_vector_type(16))) __bf16 v16bf;
typedef __attribute__((ext_vector_type(8)))  float  v8f;

union FragBF {
    v16bf v;
    unsigned short s[16];
};

// pack two f32 into a bf16 pair (round-half-up) with one v_perm_b32
__device__ __forceinline__ unsigned int pack_bf16_2(float x, float y) {
    unsigned int ux = __float_as_uint(x) + 0x8000u;
    unsigned int uy = __float_as_uint(y) + 0x8000u;
    // D.b0=S1.b2, D.b1=S1.b3, D.b2=S0.b2, D.b3=S0.b3 -> {hi16(y), hi16(x)}
    return __builtin_amdgcn_perm(uy, ux, 0x07060302u);
}

// Init: out[r] = out_b[selector[r]]  (atomic partials accumulate on top)
__global__ __launch_bounds__(256)
void nnue_init_out(const int* __restrict__ selectors,
                   const float* __restrict__ out_b,
                   float* __restrict__ out, int n) {
    int i = blockIdx.x * blockDim.x + threadIdx.x;
    if (i < n) out[i] = out_b[selectors[i]];
}

// Fused: combined-tile GEMM (bf16 WMMA, f32 acc) + bias + clip(0,6)
//        + bucketed output dot, atomically accumulated into out[row].
// Block tile 128 rows x 256 cols; 8 waves, each owning a 64x64 tile
// (4x4 subtiles -> 16 WMMAs per K-step, 8 fragment loads).
// Double-buffered LDS + register prefetch of the next K tile.
__global__ __launch_bounds__(256)
void nnue_fused_gemm(const float* __restrict__ stm,
                     const float* __restrict__ nstm,
                     const int*   __restrict__ selectors,
                     const float* __restrict__ acc_w,
                     const float* __restrict__ acc_b,
                     const float* __restrict__ out_w,
                     float* __restrict__ out)
{
    __shared__ unsigned int sA[2][BROWS][KT2];   // embeddings bf16 pairs, 16 KB
    __shared__ unsigned int sB[2][BCOLS][KT2];   // acc_w      bf16 pairs, 32 KB

    const int tid  = threadIdx.x;
    const int lane = tid & 31;
    const int wid  = tid >> 5;          // 8 waves
    const int wave_row = wid & 1;       // 2 groups x 64 rows
    const int wave_col = wid >> 1;      // 4 groups x 64 cols

    const int cb       = blockIdx.x;    // 0..7
    const int is_nstm  = cb >> 2;
    const int colW0    = (cb & 3) * BCOLS;        // weight-column base
    const int colC_off = is_nstm ? ACC_SIZE : 0;  // offset into combined/out_w
    const int row0     = blockIdx.y * BROWS;

    const float* __restrict__ emb = is_nstm ? nstm : stm;

#if defined(__HIP_DEVICE_COMPILE__)
    const int m  = lane & 15;           // row/col within a 16x16 subtile
    const int hb = lane >> 4;           // wave half (K-group select)

    // per-thread staging coordinates (A: 4 float4, B: 8 float4 per K-step)
    int rA[4], cA[4], rB[8], cB[8];     // c* = u32-pair column (0..15, even)
    #pragma unroll
    for (int p = 0; p < 4; ++p) {
        const int idx = tid + p * 256;  // 0..1023
        rA[p] = idx >> 3;               // 0..127
        cA[p] = (idx & 7) * 2;          // uint2 slot
    }
    #pragma unroll
    for (int p = 0; p < 8; ++p) {
        const int idx = tid + p * 256;  // 0..2047
        rB[p] = idx >> 3;               // 0..255
        cB[p] = (idx & 7) * 2;
    }

    // issue global loads of K-tile kt into registers
    auto loadG = [&](int kt, float4* ra, float4* rb) {
        const int K0 = kt * KT;
        #pragma unroll
        for (int p = 0; p < 4; ++p)
            ra[p] = *(const float4*)(&emb[(size_t)(row0 + rA[p]) * INPUT_SIZE + K0 + cA[p] * 2]);
        #pragma unroll
        for (int p = 0; p < 8; ++p)
            rb[p] = *(const float4*)(&acc_w[(size_t)(colW0 + rB[p]) * INPUT_SIZE + K0 + cB[p] * 2]);
    };
    // convert (perm-pack) + store a staged K-tile into LDS buffer `buf`
    auto storeL = [&](int buf, const float4* ra, const float4* rb) {
        #pragma unroll
        for (int p = 0; p < 4; ++p) {
            uint2 w;
            w.x = pack_bf16_2(ra[p].x, ra[p].y);
            w.y = pack_bf16_2(ra[p].z, ra[p].w);
            *(uint2*)&sA[buf][rA[p]][cA[p]] = w;   // one ds_store_b64
        }
        #pragma unroll
        for (int p = 0; p < 8; ++p) {
            uint2 w;
            w.x = pack_bf16_2(rb[p].x, rb[p].y);
            w.y = pack_bf16_2(rb[p].z, rb[p].w);
            *(uint2*)&sB[buf][rB[p]][cB[p]] = w;
        }
    };

    v8f acc[4][4];
    #pragma unroll
    for (int i = 0; i < 4; ++i)
        #pragma unroll
        for (int j = 0; j < 4; ++j)
            acc[i][j] = v8f{0.f, 0.f, 0.f, 0.f, 0.f, 0.f, 0.f, 0.f};

    // fragment loads (contiguous k = 16*hb + s for A and B) + 16 WMMAs on buffer `cur`
    auto compute = [&](int cur) {
        // A 16x32: lanes 0-15 hold K 0..15; lanes 16-31: K 16..31
        FragBF a[4];
        #pragma unroll
        for (int i = 0; i < 4; ++i) {
            const unsigned short* src =
                (const unsigned short*)&sA[cur][wave_row * 64 + i * 16 + m][0];
            const int kb = hb * 16;
            #pragma unroll
            for (int q = 0; q < 8; ++q) {
                a[i].s[q]     = src[kb + q];
                a[i].s[8 + q] = src[kb + 8 + q];
            }
        }
        // B 32x16: lanes 0-15 hold K 0..15 of col n; lanes 16-31 hold K 16..31
        FragBF b[4];
        #pragma unroll
        for (int j = 0; j < 4; ++j) {
            const unsigned short* src =
                (const unsigned short*)&sB[cur][wave_col * 64 + j * 16 + m][0] + hb * 16;
            #pragma unroll
            for (int q = 0; q < 16; ++q) b[j].s[q] = src[q];
        }
        #pragma unroll
        for (int i = 0; i < 4; ++i)
            #pragma unroll
            for (int j = 0; j < 4; ++j)
                acc[i][j] = __builtin_amdgcn_wmma_f32_16x16x32_bf16(
                    /*neg_a=*/false, a[i].v, /*neg_b=*/false, b[j].v,
                    /*c_mod=*/(short)0, acc[i][j],
                    /*reuse_a=*/false, /*reuse_b=*/false);
    };

    // prologue: stage K-tile 0
    {
        float4 ra[4], rb[8];
        loadG(0, ra, rb);
        storeL(0, ra, rb);
    }

    // steady state: loads for kt+1 fly under the WMMAs on kt; stores of kt+1
    // target the buffer whose last readers finished before this barrier
    for (int kt = 0; kt < NKT - 1; ++kt) {
        __syncthreads();
        const int cur = kt & 1;

        float4 ra[4], rb[8];
        loadG(kt + 1, ra, rb);
        if (kt + 2 < NKT) {             // L2 prefetch of the K+2 tile
            const int K2 = (kt + 2) * KT;
            __builtin_prefetch(&emb[(size_t)(row0 + rA[0]) * INPUT_SIZE + K2], 0, 1);
            __builtin_prefetch(&acc_w[(size_t)(colW0 + rB[0]) * INPUT_SIZE + K2], 0, 1);
        }

        compute(cur);
        storeL(cur ^ 1, ra, rb);
    }
    // last K-step (peeled: no staging)
    __syncthreads();
    compute((NKT - 1) & 1);

    // ---- epilogue: bias + clip(0,6) + bucket dot + atomic accumulate ----
    // C layout: VGPR jj -> M = jj + 8*hb, N = m
    float bias[4];
    #pragma unroll
    for (int j = 0; j < 4; ++j)
        bias[j] = acc_b[colW0 + wave_col * 64 + j * 16 + m];

    #pragma unroll
    for (int i = 0; i < 4; ++i) {
        #pragma unroll
        for (int jj = 0; jj < 8; ++jj) {
            const int rowg = row0 + wave_row * 64 + i * 16 + hb * 8 + jj;
            const int sel  = selectors[rowg];
            const float* __restrict__ w = &out_w[(size_t)sel * COMBINED + colC_off];
            float p = 0.f;
            #pragma unroll
            for (int j = 0; j < 4; ++j) {
                const int colW = colW0 + wave_col * 64 + j * 16 + m;
                float v = acc[i][j][jj] + bias[j];
                v = fminf(fmaxf(v, 0.f), 6.f);
                p += v * w[colW];
            }
            // reduce across the 16 lanes holding this row (per wave-half)
            #pragma unroll
            for (int off = 8; off >= 1; off >>= 1)
                p += __shfl_xor(p, off, 16);
            if (m == 0) atomicAdd(&out[rowg], p);
        }
    }
#endif
}

extern "C" void kernel_launch(void* const* d_in, const int* in_sizes, int n_in,
                              void* d_out, int out_size, void* d_ws, size_t ws_size,
                              hipStream_t stream) {
    const float* stm    = (const float*)d_in[0];
    const float* nstm   = (const float*)d_in[1];
    const int*   sel    = (const int*)  d_in[2];
    const float* acc_w  = (const float*)d_in[3];
    const float* acc_b  = (const float*)d_in[4];
    const float* out_w  = (const float*)d_in[5];
    const float* out_b  = (const float*)d_in[6];
    float*       out    = (float*)d_out;

    const int B = in_sizes[0] / INPUT_SIZE;   // 16384

    nnue_init_out<<<(B + 255) / 256, 256, 0, stream>>>(sel, out_b, out, B);

    dim3 grid(2 * (ACC_SIZE / BCOLS), B / BROWS);   // (8, 128)
    nnue_fused_gemm<<<grid, 256, 0, stream>>>(stm, nstm, sel, acc_w, acc_b,
                                              out_w, out);
}
